// LSTM_regressor_49254684950852
// MI455X (gfx1250) — matrix operand order repaired and checked by reading, not compiled, not run on hardware
//
#include <hip/hip_runtime.h>
#include <hip/hip_bf16.h>
#include <math.h>

// ---------------------------------------------------------------------------
// LSTM step, fused:  gates = x@Wx^T + b + h@Wh^T  (bf16 WMMA, f32 accum)
//                    c' = f*c + i*ct ; h' = o*tanh(c') ; y = h'@Wy^T + by
// Memory-bound (~486 MB min HBM traffic vs ~687 GFLOP @ 23.3 TB/s ~ 21us).
// Pipelined LDS-staged GEMM: per K-step the block cooperatively stages the
// A tile (256x32) and W tile (4x32x32) as bf16 in LDS (converted once), waves
// consume fragments via ds_load while the next tile's global loads are in
// flight. Double-buffered, one barrier per K-step.
// Block tile 256(b) x 32(h) x 4 gates; wave tile 64x16x4 (128 acc VGPRs).
// ---------------------------------------------------------------------------

typedef __attribute__((ext_vector_type(16))) __bf16 v16bf;
typedef __attribute__((ext_vector_type(8)))  __bf16 v8bf;
typedef __attribute__((ext_vector_type(4)))  __bf16 v4bf;
typedef __attribute__((ext_vector_type(8)))  float  v8f;

#define IN_DIM  1024
#define H_DIM   4096
#define B_DIM   4096
#define HTILES  (H_DIM / 16)       // 256
#define WM      4                  // 16-row m-subtiles per wave (64 b rows)
#define BLK_M   256                // 4 m-waves * 64 rows
#define BLK_N   32                 // 2 n-waves * 16 cols
#define XKT     (IN_DIM / 32)      // 32  K-steps in the x-projection
#define NT      ((IN_DIM + H_DIM) / 32)  // 160 total K-steps

__device__ __forceinline__ __bf16 f2bf(float f) {
    return static_cast<__bf16>(f);           // v_cvt_pk_bf16_f32 when paired
}

__device__ __forceinline__ float sigmoidf_(float x) {
    return 1.0f / (1.0f + __expf(-x));
}

__device__ __forceinline__ v8f wmma_bf16(v16bf a, v16bf b, v8f c) {
    // (neg_a, A, neg_b, B, c_mod, C, reuse_a, reuse_b)
    return __builtin_amdgcn_wmma_f32_16x16x32_bf16(
        false, a, false, b, (short)0, c, false, false);
}

__global__ __launch_bounds__(256)
void lstm_gates_kernel(const float* __restrict__ x,
                       const float* __restrict__ hidden,
                       const float* __restrict__ cell,
                       const float* __restrict__ Wxi, const float* __restrict__ Wxf,
                       const float* __restrict__ Wxo, const float* __restrict__ Wxc,
                       const float* __restrict__ bxi, const float* __restrict__ bxf,
                       const float* __restrict__ bxo, const float* __restrict__ bxc,
                       const float* __restrict__ Whi, const float* __restrict__ Whf,
                       const float* __restrict__ Who, const float* __restrict__ Whc,
                       const float* __restrict__ Wy,
                       float* __restrict__ ypart) {
    // LDS tiles, double-buffered. A: 256 rows x 32 K bf16 (16KB). W: 4 gates x
    // 32 rows x 32 K bf16 (8KB).
    __shared__ __attribute__((aligned(64))) __bf16 Abuf[2][BLK_M * 32];
    __shared__ __attribute__((aligned(64))) __bf16 Wbuf[2][4 * BLK_N * 32];

    const int t    = threadIdx.x;
    const int lane = t & 31;
    const int wave = t >> 5;
    const int wm   = wave & 3;               // m position (0..3)
    const int wn   = wave >> 2;              // n position (0..1)
    const int lrow = lane & 15;
    const int kg   = lane >> 4;              // K-group select within fragments

    const int bblk = blockIdx.x * BLK_M;
    const int hblk = blockIdx.y * BLK_N;
    const int bbase = bblk + wm * (WM * 16);
    const int htile = blockIdx.y * 2 + wn;
    const int h0    = htile * 16;

    const float* WxArr[4] = { Wxi, Wxf, Wxo, Wxc };
    const float* WhArr[4] = { Whi, Whf, Who, Whc };

    // ---- staging registers for the in-flight (next) tile ----
    float4 areg[8];
    float4 wreg[4];

    // Issue global loads for K-step `kt` into staging registers.
    auto issue = [&](int kt) {
#pragma unroll
        for (int j = 0; j < 8; ++j) {
            const int idx = j * 256 + t;
            const int row = idx >> 3;        // 0..255
            const int col = idx & 7;         // float4 column
            const int ar  = bblk + row;
            const float* src = (kt < XKT)
                ? x      + (size_t)ar * IN_DIM + kt * 32 + col * 4
                : hidden + (size_t)ar * H_DIM + (kt - XKT) * 32 + col * 4;
            areg[j] = *(const float4*)src;
        }
        const int row = t >> 3;              // 0..31
        const int col = t & 7;
        const int hr  = hblk + row;
#pragma unroll
        for (int j = 0; j < 4; ++j) {        // one gate per j
            const float* src = (kt < XKT)
                ? WxArr[j] + (size_t)hr * IN_DIM + kt * 32 + col * 4
                : WhArr[j] + (size_t)hr * H_DIM + (kt - XKT) * 32 + col * 4;
            wreg[j] = *(const float4*)src;
        }
    };

    // Convert staging registers to bf16 and store into LDS buffer `buf`.
    auto stage = [&](int buf) {
#pragma unroll
        for (int j = 0; j < 8; ++j) {
            const int idx = j * 256 + t;
            const int row = idx >> 3;
            const int col = idx & 7;
            v4bf v;
            v[0] = f2bf(areg[j].x); v[1] = f2bf(areg[j].y);
            v[2] = f2bf(areg[j].z); v[3] = f2bf(areg[j].w);
            *(v4bf*)&Abuf[buf][row * 32 + col * 4] = v;
        }
        const int row = t >> 3;
        const int col = t & 7;
#pragma unroll
        for (int j = 0; j < 4; ++j) {
            v4bf v;
            v[0] = f2bf(wreg[j].x); v[1] = f2bf(wreg[j].y);
            v[2] = f2bf(wreg[j].z); v[3] = f2bf(wreg[j].w);
            *(v4bf*)&Wbuf[buf][(j * BLK_N + row) * 32 + col * 4] = v;
        }
    };

    v8f acc[4][WM];
#pragma unroll
    for (int g = 0; g < 4; ++g)
#pragma unroll
        for (int m = 0; m < WM; ++m)
            acc[g][m] = (v8f){};

    // Consume LDS buffer `buf`: read fragments, 16 WMMAs.
    auto compute = [&](int buf) {
        const __bf16* Ab = Abuf[buf];
        const __bf16* Wb = Wbuf[buf];
        v16bf a[WM];
#pragma unroll
        for (int m = 0; m < WM; ++m) {
            // A frag 16x32: lanes<16 -> K 0..15; lanes>=16 -> K 16..31
            const int row = wm * 64 + m * 16 + lrow;
            const v8bf lo = *(const v8bf*)&Ab[row * 32 + kg * 16];
            const v8bf hi = *(const v8bf*)&Ab[row * 32 + kg * 16 + 8];
#pragma unroll
            for (int e = 0; e < 8; ++e) { a[m][e] = lo[e]; a[m][8 + e] = hi[e]; }
        }
        const int brow = wn * 16 + lrow;
#pragma unroll
        for (int g = 0; g < 4; ++g) {
            // B frag 32x16: lane&15 = column, lanes<16 K 0..15, lanes>=16 K 16..31
            const v16bf b =
                *(const v16bf*)&Wb[(g * BLK_N + brow) * 32 + kg * 16];
#pragma unroll
            for (int m = 0; m < WM; ++m)
                acc[g][m] = wmma_bf16(a[m], b, acc[g][m]);
        }
    };

    // ---- pipelined main loop: one barrier per K-step ----
    issue(0);
    stage(0);
    __syncthreads();
    for (int kt = 0; kt < NT; ++kt) {
        const int buf = kt & 1;
        if (kt + 1 < NT) issue(kt + 1);      // next tile's loads in flight
        compute(buf);                        // WMMAs on current LDS buffer
        if (kt + 1 < NT) stage(buf ^ 1);     // cvt + ds_store next buffer
        __syncthreads();
    }

    // ---- fused gate math + head partial reduction ----
    // C/D layout: VGPR r, lanes<16 -> (M=r, N=lane), lanes>=16 -> (M=8+r, N=lane-16)
    const int h  = h0 + lrow;
    const float bi = bxi[h], bf_ = bxf[h], bo = bxo[h], bc = bxc[h];
    const float wy = Wy[h];
    const int mhi = (lane >> 4) << 3;

#pragma unroll
    for (int m = 0; m < WM; ++m) {
#pragma unroll
        for (int r = 0; r < 8; ++r) {
            const int brow = bbase + m * 16 + r + mhi;
            const float cv = cell[(size_t)brow * H_DIM + h];
            const float ig = sigmoidf_(acc[0][m][r] + bi);
            const float fg = sigmoidf_(acc[1][m][r] + bf_);
            const float og = sigmoidf_(acc[2][m][r] + bo);
            const float ct = tanhf(acc[3][m][r] + bc);
            const float cn = fg * cv + ig * ct;
            const float hn = og * tanhf(cn);
            float p = hn * wy;
            // reduce over N (16 lanes of this half-wave)
            p += __shfl_xor(p, 1, 32);
            p += __shfl_xor(p, 2, 32);
            p += __shfl_xor(p, 4, 32);
            p += __shfl_xor(p, 8, 32);
            if (lrow == 0)
                ypart[(size_t)brow * HTILES + htile] = p;
        }
    }
}

// Deterministic fixed-order reduction of the head partials.
__global__ __launch_bounds__(256)
void lstm_head_kernel(const float* __restrict__ ypart,
                      const float* __restrict__ by,
                      float* __restrict__ y) {
    const int b = blockIdx.x * blockDim.x + threadIdx.x;
    if (b >= B_DIM) return;
    float s = by[0];
    const float* row = ypart + (size_t)b * HTILES;
#pragma unroll 8
    for (int t = 0; t < HTILES; ++t) s += row[t];
    y[b] = s;
}

extern "C" void kernel_launch(void* const* d_in, const int* in_sizes, int n_in,
                              void* d_out, int out_size, void* d_ws, size_t ws_size,
                              hipStream_t stream) {
    const float* x      = (const float*)d_in[0];
    const float* hidden = (const float*)d_in[1];
    const float* cell   = (const float*)d_in[2];
    const float* Wxi = (const float*)d_in[3];
    const float* Wxf = (const float*)d_in[4];
    const float* Wxo = (const float*)d_in[5];
    const float* Wxc = (const float*)d_in[6];
    const float* bxi = (const float*)d_in[7];
    const float* bxf = (const float*)d_in[8];
    const float* bxo = (const float*)d_in[9];
    const float* bxc = (const float*)d_in[10];
    const float* Whi = (const float*)d_in[11];
    const float* Whf = (const float*)d_in[12];
    const float* Who = (const float*)d_in[13];
    const float* Whc = (const float*)d_in[14];
    const float* Wy  = (const float*)d_in[15];
    const float* by  = (const float*)d_in[16];

    float* ypart = (float*)d_ws;   // B_DIM * HTILES floats = 4 MB

    dim3 grid(B_DIM / BLK_M, H_DIM / BLK_N);   // (16, 128)
    lstm_gates_kernel<<<grid, 256, 0, stream>>>(
        x, hidden, cell,
        Wxi, Wxf, Wxo, Wxc,
        bxi, bxf, bxo, bxc,
        Whi, Whf, Who, Whc,
        Wy, ypart);

    lstm_head_kernel<<<(B_DIM + 255) / 256, 256, 0, stream>>>(
        ypart, by, (float*)d_out);
}
